// Model_89730456748695
// MI455X (gfx1250) — compile-verified, matrix operand order checked by reading, not run on hardware
//
#include <hip/hip_runtime.h>
#include <math.h>

// ---------------- CDNA5 WMMA types ----------------
typedef __attribute__((ext_vector_type(16))) __bf16 v16bf;
typedef __attribute__((ext_vector_type(8)))  __bf16 v8bf;
typedef __attribute__((ext_vector_type(8)))  float  v8f;
typedef __attribute__((ext_vector_type(4)))  float  vf4;

#define DM   256   // d_model
#define DFF  512
#define NH   8
#define DK   32    // d_model / heads
#define BN   2
#define LENC 2048
#define LDEC 1024
#define CIN  21
#define CPAD 32
#define COUT 21
#define PRED 512

static inline int cdiv(int a, int b) { return (a + b - 1) / b; }

__device__ __forceinline__ float gelu_exact(float x) {
  return 0.5f * x * (1.0f + erff(x * 0.70710678118654752f));
}

__device__ __forceinline__ v16bf cat8(v8bf lo, v8bf hi) {
  return __builtin_shufflevector(lo, hi, 0, 1, 2, 3, 4, 5, 6, 7,
                                 8, 9, 10, 11, 12, 13, 14, 15);
}

// Build a 16x32 bf16 A-fragment from 4 float4 row loads (groups are the
// contiguous K-ranges [k0+half*8, +8) and [k0+16+half*8, +8)).
__device__ __forceinline__ v16bf afrag_f32(vf4 a0, vf4 a1, vf4 a2, vf4 a3) {
  v16bf a;
#pragma unroll
  for (int i = 0; i < 4; ++i) {
    a[i]      = (__bf16)a0[i];
    a[4 + i]  = (__bf16)a1[i];
    a[8 + i]  = (__bf16)a2[i];
    a[12 + i] = (__bf16)a3[i];
  }
  return a;
}

// ---------------------------------------------------------------------------
// GEMM:  C[M,N] = act( A'[M,Kpad] @ Wt^T + bias ) (+ addsrc) (+= C)
// Wt is pre-transposed bf16 [Npad][Kpad] (zero padded).  Kpad%32==0,
// Npad%64==0.  One wave computes a 16x64 tile (4 WMMAs / K-chunk, shared A).
// A' row r maps to row (b, (l+rowShift) mod rowMod) when rowMod>0 (circular
// width-3 conv as 3 accumulated GEMM passes).  Hot loop is branch-free.
// ---------------------------------------------------------------------------
__global__ __launch_bounds__(32) void gemm_wmma_k(
    const float* __restrict__ A, const __bf16* __restrict__ Wt,
    const float* __restrict__ bias, const float* __restrict__ addsrc,
    float* __restrict__ C, int M, int N, int Kpad, int lda, int ldc,
    int rowShift, int rowMod, int act, int accumulate)
{
  const int lane = threadIdx.x & 31;
  const int half = lane >> 4;
  const int l15  = lane & 15;
  const int mtile = blockIdx.x;
  const int nbase = blockIdx.y * 64;

  int arow = mtile * 16 + l15;
  if (rowMod > 0) {
    const int bb = arow / rowMod;
    int ll = arow % rowMod;
    ll = (ll + rowShift + rowMod) % rowMod;
    arow = bb * rowMod + ll;
  }
  const vf4* __restrict__ Ap = (const vf4*)(A + (size_t)arow * lda);

  v8f acc[4] = {};
  for (int k0 = 0; k0 < Kpad; k0 += 32) {
    const int base = (k0 >> 2) + half * 2;
    const v16bf a = afrag_f32(Ap[base], Ap[base + 1], Ap[base + 4], Ap[base + 5]);
#pragma unroll
    for (int nt = 0; nt < 4; ++nt) {
      const __bf16* __restrict__ wn = Wt + (size_t)(nbase + nt * 16 + l15) * Kpad;
      const v16bf b = cat8(*(const v8bf*)(wn + k0 + half * 8),
                           *(const v8bf*)(wn + k0 + 16 + half * 8));
      acc[nt] = __builtin_amdgcn_wmma_f32_16x16x32_bf16(
          false, a, false, b, (short)0, acc[nt], false, false);
    }
  }

#pragma unroll
  for (int nt = 0; nt < 4; ++nt) {
#pragma unroll
    for (int r = 0; r < 8; ++r) {
      const int m = mtile * 16 + r + half * 8;
      const int n = nbase + nt * 16 + l15;
      if (n < N) {
        float val = acc[nt][r];
        if (bias)       val += bias[n];
        if (accumulate) val += C[(size_t)m * ldc + n];
        if (addsrc)     val += addsrc[(size_t)m * ldc + n];
        if (act == 1)   val = gelu_exact(val);
        C[(size_t)m * ldc + n] = val;
      }
    }
  }
}

// ---------------------------------------------------------------------------
// Flash-style streaming attention, one wave per (b, h, 16-row Q tile).
// QK^T via WMMA (Q row-major float4 loads), online softmax with half-wave
// shuffles, P staged through LDS (C-layout -> A-fragment), P@V via WMMA with
// V pre-transposed to bf16 [b][h][dk][L] (contiguous b128 fragment loads).
// LQ, LK are multiples of 32 -> hot loop branch-free except causal VALU mask.
// ---------------------------------------------------------------------------
__global__ __launch_bounds__(32) void attn_wmma_k(
    const float* __restrict__ Q, const float* __restrict__ Km,
    const __bf16* __restrict__ Vt, float* __restrict__ O,
    int LQ, int LK, int causal, float scale)
{
  const int lane = threadIdx.x & 31;
  const int half = lane >> 4;
  const int l15  = lane & 15;
  const int qbase = blockIdx.x * 16;
  const int h = blockIdx.y;
  const int b = blockIdx.z;

  const float* Qb  = Q  + ((size_t)b * LQ) * DM + h * DK;
  const float* Kb  = Km + ((size_t)b * LK) * DM + h * DK;
  const __bf16* Vb = Vt + ((size_t)(b * NH + h) * DK) * LK;
  float*       Ob  = O  + ((size_t)b * LQ) * DM + h * DK;

  __shared__ __bf16 pshm[16 * 32];

  // Q fragment (16x32 bf16), pre-scaled by 1/sqrt(dk)
  const vf4* Qp = (const vf4*)(Qb + (size_t)(qbase + l15) * DM);
  const v16bf aq = afrag_f32(Qp[half * 2] * scale, Qp[half * 2 + 1] * scale,
                             Qp[4 + half * 2] * scale, Qp[5 + half * 2] * scale);

  v8f o0 = {}, o1 = {};
  float mrow[8], lrow[8];
#pragma unroll
  for (int r = 0; r < 8; ++r) { mrow[r] = -__builtin_inff(); lrow[r] = 0.0f; }

  for (int kt = 0; kt < LK; kt += 32) {
    if (causal && kt > qbase + 15) break;

    // two 16x16 score tiles (columns kt..kt+15, kt+16..kt+31)
    v8f s[2];
#pragma unroll
    for (int nt = 0; nt < 2; ++nt) {
      const int kcol = kt + nt * 16 + l15;     // K-row index = score column
      const vf4* Kp = (const vf4*)(Kb + (size_t)kcol * DM);
      const v16bf bk = afrag_f32(Kp[half * 2], Kp[half * 2 + 1],
                                 Kp[4 + half * 2], Kp[5 + half * 2]);
      v8f z = {};
      s[nt] = __builtin_amdgcn_wmma_f32_16x16x32_bf16(false, aq, false, bk,
                                                      (short)0, z, false, false);
    }

    // online softmax per row (row r+half*8 owned by this half-wave)
#pragma unroll
    for (int r = 0; r < 8; ++r) {
      const int row = qbase + r + half * 8;
      float s0 = s[0][r], s1 = s[1][r];
      const bool k0v = !causal || (kt + l15) <= row;
      const bool k1v = !causal || (kt + 16 + l15) <= row;
      if (!k0v) s0 = -__builtin_inff();
      if (!k1v) s1 = -__builtin_inff();
      float cm = fmaxf(s0, s1);
      for (int i = 1; i < 16; i <<= 1) cm = fmaxf(cm, __shfl_xor(cm, i, 16));
      const float mnew = fmaxf(mrow[r], cm);
      const float p0 = k0v ? __expf(s0 - mnew) : 0.0f;
      const float p1 = k1v ? __expf(s1 - mnew) : 0.0f;
      float rs = p0 + p1;
      for (int i = 1; i < 16; i <<= 1) rs += __shfl_xor(rs, i, 16);
      const float alpha =
          (mrow[r] == -__builtin_inff()) ? 0.0f : __expf(mrow[r] - mnew);
      lrow[r] = lrow[r] * alpha + rs;
      o0[r] *= alpha; o1[r] *= alpha;
      mrow[r] = mnew;
      pshm[(r + half * 8) * 32 + l15]      = (__bf16)p0;
      pshm[(r + half * 8) * 32 + 16 + l15] = (__bf16)p1;
    }

    // reload P (16x32) as A fragment (same-wave DS ops are in-order)
    const __bf16* pr = &pshm[l15 * 32];
    const v16bf pa = cat8(*(const v8bf*)(pr + half * 8),
                          *(const v8bf*)(pr + 16 + half * 8));

    // ctx accumulation over two 16-wide column tiles of V (transposed bf16)
#pragma unroll
    for (int vt = 0; vt < 2; ++vt) {
      const __bf16* vn = Vb + (size_t)(vt * 16 + l15) * LK;
      const v16bf bv = cat8(*(const v8bf*)(vn + kt + half * 8),
                            *(const v8bf*)(vn + kt + 16 + half * 8));
      if (vt == 0)
        o0 = __builtin_amdgcn_wmma_f32_16x16x32_bf16(false, pa, false, bv,
                                                     (short)0, o0, false, false);
      else
        o1 = __builtin_amdgcn_wmma_f32_16x16x32_bf16(false, pa, false, bv,
                                                     (short)0, o1, false, false);
    }
  }

#pragma unroll
  for (int r = 0; r < 8; ++r) {
    const int row = qbase + r + half * 8;
    const float inv = (lrow[r] > 0.0f) ? (1.0f / lrow[r]) : 0.0f;
    Ob[(size_t)row * DM + l15]      = o0[r] * inv;
    Ob[(size_t)row * DM + 16 + l15] = o1[r] * inv;
  }
}

// ---------------------------------------------------------------------------
// Weight prep: W[K,N] f32 (row stride ldw) -> Wt[Npad][Kpad] bf16, zero pad.
// ---------------------------------------------------------------------------
__global__ void prep_w_k(const float* __restrict__ W, __bf16* __restrict__ Wt,
                         int K, int N, int ldw, int Kpad, int total)
{
  int i = blockIdx.x * blockDim.x + threadIdx.x;
  if (i >= total) return;
  const int k = i % Kpad;
  const int n = i / Kpad;
  Wt[i] = (k < K && n < N) ? (__bf16)W[(size_t)k * ldw + n] : (__bf16)0.0f;
}

// V (B,L,DM) f32 -> Vt bf16 [b][h][dk][L]
__global__ void transpose_v_k(const float* __restrict__ V,
                              __bf16* __restrict__ Vt, int L, int total)
{
  int i = blockIdx.x * blockDim.x + threadIdx.x;
  if (i >= total) return;
  const int c = i % DM;
  const int l = (i / DM) % L;
  const int b = i / (DM * L);
  const int h = c / DK, n = c % DK;
  Vt[((size_t)(b * NH + h) * DK + n) * L + l] = (__bf16)V[i];
}

// pad channels 21->32, optional instance-norm by per-(b,c) mean/std
__global__ void pad_norm_k(const float* __restrict__ x, const float* mean,
                           const float* stdv, float* __restrict__ y,
                           int L, int Cin, int Cpad, int donorm, int total)
{
  int i = blockIdx.x * blockDim.x + threadIdx.x;
  if (i >= total) return;
  const int c = i % Cpad;
  const int bl = i / Cpad;
  const int b = bl / L;
  float v = 0.0f;
  if (c < Cin) {
    v = x[(size_t)bl * Cin + c];
    if (donorm) v = (v - mean[b * Cin + c]) / stdv[b * Cin + c];
  }
  y[i] = v;
}

// LayerNorm over last dim, one wave per row
__global__ __launch_bounds__(32) void layernorm_k(
    const float* __restrict__ X, const float* __restrict__ g,
    const float* __restrict__ bta, float* __restrict__ Y, int N)
{
  const int row = blockIdx.x;
  const int lane = threadIdx.x;
  const float* x = X + (size_t)row * N;
  float* y = Y + (size_t)row * N;
  float s = 0.f, s2 = 0.f;
  for (int c = lane; c < N; c += 32) { float v = x[c]; s += v; s2 += v * v; }
  for (int i = 1; i < 32; i <<= 1) { s += __shfl_xor(s, i, 32); s2 += __shfl_xor(s2, i, 32); }
  const float mean = s / N;
  const float inv = rsqrtf(s2 / N - mean * mean + 1e-5f);
  for (int c = lane; c < N; c += 32)
    y[c] = (x[c] - mean) * inv * g[c] + bta[c];
}

// per-(b,channel) mean/std over sequence axis
__global__ __launch_bounds__(32) void stats_k(
    const float* __restrict__ x, float* mean, float* stdv, int L, int Cc)
{
  const int bc = blockIdx.x, b = bc / Cc, c = bc % Cc;
  const int lane = threadIdx.x;
  float s = 0.f, s2 = 0.f;
  for (int l = lane; l < L; l += 32) {
    float v = x[((size_t)b * L + l) * Cc + c];
    s += v; s2 += v * v;
  }
  for (int i = 1; i < 32; i <<= 1) { s += __shfl_xor(s, i, 32); s2 += __shfl_xor(s2, i, 32); }
  if (lane == 0) {
    float m = s / L;
    mean[bc] = m;
    stdv[bc] = sqrtf(s2 / L - m * m + 1e-5f);
  }
}

// sinusoidal positional embedding add
__global__ void posadd_k(float* __restrict__ X, int L, int D, int total)
{
  int i = blockIdx.x * blockDim.x + threadIdx.x;
  if (i >= total) return;
  const int c = i % D;
  const int l = (i / D) % L;
  const int ci = c & ~1;
  const float freq = __expf(-logf(10000.0f) * (float)ci / (float)D);
  const float ang = (float)l * freq;
  X[i] += (c & 1) ? __cosf(ang) : __sinf(ang);
}

// distil tail: y = elu(conv/sqrt(1+eps)*g + b) then maxpool(3, stride 2, pad 1)
__global__ void distilpost_k(const float* __restrict__ cbuf,
                             const float* __restrict__ g,
                             const float* __restrict__ bb,
                             float* __restrict__ out,
                             int Lin, int Lout, int D, int total)
{
  int i = blockIdx.x * blockDim.x + threadIdx.x;
  if (i >= total) return;
  const int c = i % D;
  const int j = (i / D) % Lout;
  const int b = (i / D) / Lout;
  const float invs = rsqrtf(1.0f + 1e-5f);
  float best = -__builtin_inff();
  for (int t = -1; t <= 1; ++t) {
    const int l = 2 * j + t;
    if (l < 0 || l >= Lin) continue;
    float v = cbuf[((size_t)b * Lin + l) * D + c] * invs * g[c] + bb[c];
    v = (v > 0.0f) ? v : (__expf(v) - 1.0f);
    best = fmaxf(best, v);
  }
  out[i] = best;
}

// final: de-normalize and keep last PRED steps
__global__ void out_k(const float* __restrict__ proj, const float* mean,
                      const float* stdv, float* __restrict__ out,
                      int Ldec, int Pred, int Cc, int total)
{
  int i = blockIdx.x * blockDim.x + threadIdx.x;
  if (i >= total) return;
  const int c = i % Cc;
  const int j = (i / Cc) % Pred;
  const int b = (i / Cc) / Pred;
  const float v = proj[((size_t)b * Ldec + (Ldec - Pred + j)) * Cc + c];
  out[i] = v * stdv[b * Cc + c] + mean[b * Cc + c];
}

// ---------------------------------------------------------------------------
extern "C" void kernel_launch(void* const* d_in, const int* in_sizes, int n_in,
                              void* d_out, int out_size, void* d_ws, size_t ws_size,
                              hipStream_t stream) {
  (void)in_sizes; (void)n_in; (void)out_size; (void)ws_size;
  const float* x_enc = (const float*)d_in[0];
  const float* x_dec = (const float*)d_in[3];

  // params flattened in dict insertion order starting at index 7
  const int P = 7;
  auto in = [&](int i) { return (const float*)d_in[i]; };
  const float* emb_enc = in(P + 0);
  const float* emb_dec = in(P + 1);
  const float* enc_ng = in(P + 2); const float* enc_nb = in(P + 3);
  const float* dec_ng = in(P + 4); const float* dec_nb = in(P + 5);
  const float* proj_w = in(P + 6); const float* proj_b = in(P + 7);
  const int ENC0 = P + 8;            // 16 leaves per encoder layer
  const int ENC1 = P + 8 + 16;
  const int CNV0 = P + 8 + 32;       // conv0: w,b,bn_g,bn_b
  const int DEC0 = CNV0 + 4;         // self(8) cross(8) ff(4) ln(6)

  // workspace carve (256B aligned slabs)
  char* wsp = (char*)d_ws;
  auto allocB = [&](size_t bytes) {
    void* p = (void*)wsp;
    wsp += ((bytes + 255) / 256) * 256;
    return p;
  };
  auto allocF = [&](size_t n) { return (float*)allocB(n * sizeof(float)); };
  float* meanb = allocF(BN * CIN);
  float* stdb  = allocF(BN * CIN);
  float* xep   = allocF((size_t)BN * LENC * CPAD);  // padded norm enc input
  float* xdp   = allocF((size_t)BN * LDEC * CPAD);  // padded dec input
  float* bufA  = allocF((size_t)BN * LENC * DM);
  float* bufB  = allocF((size_t)BN * LENC * DM);
  float* qb    = allocF((size_t)BN * LENC * DM);
  float* kbuf  = allocF((size_t)BN * LENC * DM);
  float* vbuf  = allocF((size_t)BN * LENC * DM);
  float* atb   = allocF((size_t)BN * LENC * DM);
  float* ffb   = allocF((size_t)BN * LENC * DFF);
  float* encout= allocF((size_t)BN * LDEC * DM);
  float* projb = allocF((size_t)BN * LDEC * COUT);
  __bf16* wtb  = (__bf16*)allocB((size_t)512 * 512 * sizeof(__bf16)); // Wt scratch
  __bf16* vtb  = (__bf16*)allocB((size_t)BN * NH * DK * LENC * sizeof(__bf16));

  const float scale = 0.17677669529663688f; // 1/sqrt(32)

  // prep weight then GEMM
  auto gemmW = [&](const float* A, const float* W_, int K, int N, int ldw,
                   const float* bias, const float* addsrc, float* C,
                   int M, int lda, int ldc, int rowShift, int rowMod,
                   int act, int accum) {
    const int Kpad = cdiv(K, 32) * 32;
    const int Npad = cdiv(N, 64) * 64;
    const int total = Npad * Kpad;
    prep_w_k<<<cdiv(total, 256), 256, 0, stream>>>(W_, wtb, K, N, ldw, Kpad, total);
    dim3 g(M / 16, Npad / 64);
    gemm_wmma_k<<<g, dim3(32), 0, stream>>>(A, wtb, bias, addsrc, C, M, N, Kpad,
                                            lda, ldc, rowShift, rowMod, act, accum);
  };

  auto attention = [&](const float* Qa, const float* Ka, const float* Va,
                       float* Oa, int LQv, int LKv, int causal) {
    const int total = BN * LKv * DM;
    transpose_v_k<<<cdiv(total, 256), 256, 0, stream>>>(Va, vtb, LKv, total);
    attn_wmma_k<<<dim3(LQv / 16, NH, BN), dim3(32), 0, stream>>>(
        Qa, Ka, vtb, Oa, LQv, LKv, causal, scale);
  };

  auto embed = [&](const float* srcPad, const float* wconv, float* dst, int Lcur) {
    const int M = BN * Lcur;
    for (int t = 0; t < 3; ++t)
      gemmW(srcPad, wconv + (size_t)t * CIN * DM, CIN, DM, DM, nullptr, nullptr,
            dst, M, CPAD, DM, t - 1, Lcur, 0, t > 0);
    const int total = M * DM;
    posadd_k<<<cdiv(total, 256), 256, 0, stream>>>(dst, Lcur, DM, total);
  };

  auto enc_layer = [&](float* x, int Lcur, int pb) {
    const int M = BN * Lcur;
    gemmW(x, in(pb + 0), DM, DM, DM, in(pb + 1), nullptr, qb,   M, DM, DM, 0, 0, 0, 0);
    gemmW(x, in(pb + 2), DM, DM, DM, in(pb + 3), nullptr, kbuf, M, DM, DM, 0, 0, 0, 0);
    gemmW(x, in(pb + 4), DM, DM, DM, in(pb + 5), nullptr, vbuf, M, DM, DM, 0, 0, 0, 0);
    attention(qb, kbuf, vbuf, atb, Lcur, Lcur, 0);
    gemmW(atb, in(pb + 6), DM, DM, DM, in(pb + 7), x, bufB, M, DM, DM, 0, 0, 0, 0);
    layernorm_k<<<M, 32, 0, stream>>>(bufB, in(pb + 12), in(pb + 13), x, DM);
    gemmW(x, in(pb + 8), DM, DFF, DFF, in(pb + 9), nullptr, ffb, M, DM, DFF, 0, 0, 1, 0);
    gemmW(ffb, in(pb + 10), DFF, DM, DM, in(pb + 11), x, bufB, M, DFF, DM, 0, 0, 0, 0);
    layernorm_k<<<M, 32, 0, stream>>>(bufB, in(pb + 14), in(pb + 15), x, DM);
  };

  // ---------------- encoder ----------------
  stats_k<<<BN * CIN, 32, 0, stream>>>(x_enc, meanb, stdb, LENC, CIN);
  {
    const int total = BN * LENC * CPAD;
    pad_norm_k<<<cdiv(total, 256), 256, 0, stream>>>(x_enc, meanb, stdb, xep,
                                                     LENC, CIN, CPAD, 1, total);
  }
  embed(xep, emb_enc, bufA, LENC);
  enc_layer(bufA, LENC, ENC0);

  // distil: width-3 circular conv as 3 shifted GEMMs, then elu + maxpool/2
  for (int t = 0; t < 3; ++t)
    gemmW(bufA, in(CNV0 + 0) + (size_t)t * DM * DM, DM, DM, DM,
          (t == 0) ? in(CNV0 + 1) : nullptr, nullptr, vbuf,
          BN * LENC, DM, DM, t - 1, LENC, 0, t > 0);
  {
    const int total = BN * (LENC / 2) * DM;
    distilpost_k<<<cdiv(total, 256), 256, 0, stream>>>(
        vbuf, in(CNV0 + 2), in(CNV0 + 3), bufA, LENC, LENC / 2, DM, total);
  }
  enc_layer(bufA, LENC / 2, ENC1);
  layernorm_k<<<BN * (LENC / 2), 32, 0, stream>>>(bufA, enc_ng, enc_nb, encout, DM);

  // ---------------- decoder ----------------
  {
    const int total = BN * LDEC * CPAD;
    pad_norm_k<<<cdiv(total, 256), 256, 0, stream>>>(x_dec, meanb, stdb, xdp,
                                                     LDEC, CIN, CPAD, 0, total);
  }
  embed(xdp, emb_dec, bufA, LDEC);
  {
    const int pb = DEC0, M = BN * LDEC;
    // masked self-attention
    gemmW(bufA, in(pb + 0), DM, DM, DM, in(pb + 1), nullptr, qb,   M, DM, DM, 0, 0, 0, 0);
    gemmW(bufA, in(pb + 2), DM, DM, DM, in(pb + 3), nullptr, kbuf, M, DM, DM, 0, 0, 0, 0);
    gemmW(bufA, in(pb + 4), DM, DM, DM, in(pb + 5), nullptr, vbuf, M, DM, DM, 0, 0, 0, 0);
    attention(qb, kbuf, vbuf, atb, LDEC, LDEC, 1);
    gemmW(atb, in(pb + 6), DM, DM, DM, in(pb + 7), bufA, bufB, M, DM, DM, 0, 0, 0, 0);
    layernorm_k<<<M, 32, 0, stream>>>(bufB, in(pb + 20), in(pb + 21), bufA, DM);
    // cross-attention with encoder output
    gemmW(bufA,   in(pb + 8),  DM, DM, DM, in(pb + 9),  nullptr, qb,   M, DM, DM, 0, 0, 0, 0);
    gemmW(encout, in(pb + 10), DM, DM, DM, in(pb + 11), nullptr, kbuf, BN * LDEC, DM, DM, 0, 0, 0, 0);
    gemmW(encout, in(pb + 12), DM, DM, DM, in(pb + 13), nullptr, vbuf, BN * LDEC, DM, DM, 0, 0, 0, 0);
    attention(qb, kbuf, vbuf, atb, LDEC, LDEC, 0);
    gemmW(atb, in(pb + 14), DM, DM, DM, in(pb + 15), bufA, bufB, M, DM, DM, 0, 0, 0, 0);
    layernorm_k<<<M, 32, 0, stream>>>(bufB, in(pb + 22), in(pb + 23), bufA, DM);
    // FFN
    gemmW(bufA, in(pb + 16), DM, DFF, DFF, in(pb + 17), nullptr, ffb, M, DM, DFF, 0, 0, 1, 0);
    gemmW(ffb, in(pb + 18), DFF, DM, DM, in(pb + 19), bufA, bufB, M, DFF, DM, 0, 0, 0, 0);
    layernorm_k<<<M, 32, 0, stream>>>(bufB, in(pb + 24), in(pb + 25), bufA, DM);
  }
  layernorm_k<<<BN * LDEC, 32, 0, stream>>>(bufA, dec_ng, dec_nb, bufB, DM);
  gemmW(bufB, proj_w, DM, COUT, COUT, proj_b, nullptr, projb,
        BN * LDEC, DM, COUT, 0, 0, 0, 0);
  {
    const int total = BN * PRED * COUT;
    out_k<<<cdiv(total, 256), 256, 0, stream>>>(projb, meanb, stdb,
                                                (float*)d_out, LDEC, PRED, COUT, total);
  }
}
